// BufferAttend1d_79525614452792
// MI455X (gfx1250) — compile-verified, matrix operand order checked by reading, not run on hardware
//
#include <hip/hip_runtime.h>
#include <hip/hip_bf16.h>
#include <cstddef>

// ---------------------------------------------------------------------------
// Problem constants (from reference): B=4, Q=4096, K=4096, D=512, KD=VD=64
// ---------------------------------------------------------------------------
#define NB   4
#define NQ   4096
#define NK   4096
#define ND   512
#define HD   64              // head dim (KD == VD == 64)
#define LOG2E  1.4426950408889634f
#define FILL2  (-1024.0f * LOG2E)     // mask fill, pre-scaled to exp2 domain
#define SCALE2 (0.125f * LOG2E)       // (1/sqrt(64)) * log2(e)

typedef __attribute__((ext_vector_type(16))) __bf16 v16bf;
typedef __attribute__((ext_vector_type(8)))  __bf16 v8bf;
typedef __attribute__((ext_vector_type(4)))  __bf16 v4bf;
typedef __attribute__((ext_vector_type(2)))  __bf16 v2bf;
typedef __attribute__((ext_vector_type(8)))  float  v8f;

// packed f32x2 -> bf16x2: round-bias then one v_perm_b32 picks the two
// high halves (1.5 VALU/element)
static __device__ __forceinline__ v2bf cvt2(float lo, float hi) {
    unsigned ul = __builtin_bit_cast(unsigned, lo) + 0x8000u;
    unsigned uh = __builtin_bit_cast(unsigned, hi) + 0x8000u;
    return __builtin_bit_cast(v2bf, __builtin_amdgcn_perm(uh, ul, 0x07060302u));
}

static __device__ __forceinline__ __bf16 cvt1(float f) {
    unsigned u = __builtin_bit_cast(unsigned, f) + 0x8000u;
    return __builtin_bit_cast(v2bf, __builtin_amdgcn_perm(u, u, 0x07060302u))[0];
}

// load 8 consecutive f32 and convert to bf16 vector (4 packed converts)
static __device__ __forceinline__ v8bf cvt8(const float* __restrict__ p) {
    float4 a = *(const float4*)p;
    float4 b = *(const float4*)(p + 4);
    v2bf r0 = cvt2(a.x, a.y), r1 = cvt2(a.z, a.w);
    v2bf r2 = cvt2(b.x, b.y), r3 = cvt2(b.z, b.w);
    v4bf lo = __builtin_shufflevector(r0, r1, 0, 1, 2, 3);
    v4bf hi = __builtin_shufflevector(r2, r3, 0, 1, 2, 3);
    return __builtin_shufflevector(lo, hi, 0, 1, 2, 3, 4, 5, 6, 7);
}

static __device__ __forceinline__ v16bf cat16(v8bf lo, v8bf hi) {
    return __builtin_shufflevector(lo, hi, 0,1,2,3,4,5,6,7,8,9,10,11,12,13,14,15);
}

// 16-byte fragment halves from a bf16 row (LDS or global)
static __device__ __forceinline__ v16bf frag_row(const __bf16* p, int off1, int off2) {
    v8bf lo = *(const v8bf*)(p + off1);
    v8bf hi = *(const v8bf*)(p + off2);
    return cat16(lo, hi);
}

static __device__ __forceinline__ v8f wmma_bf16(v16bf a, v16bf b, v8f c) {
    // (neg_a, A, neg_b, B, c_mod, C, reuse_a, reuse_b)
    return __builtin_amdgcn_wmma_f32_16x16x32_bf16(false, a, false, b, (short)0, c,
                                                   false, false);
}

// ---------------------------------------------------------------------------
// Kernel 0: one-shot weight conversion Wq/Wk/Wv f32 -> bf16 workspace.
// 3 * 64*512 elements; grid = 48 blocks * 256 threads * 8 elem/thread.
// ---------------------------------------------------------------------------
__global__ __launch_bounds__(256)
void wcvt_kernel(const float* __restrict__ Wq, const float* __restrict__ Wk,
                 const float* __restrict__ Wv, __bf16* __restrict__ wdst) {
    const int per = (HD * ND) / (256 * 8);            // 16 blocks per matrix
    const int which = blockIdx.x / per;
    const float* W = (which == 0) ? Wq : (which == 1) ? Wk : Wv;
    const int off = ((blockIdx.x % per) * 256 + threadIdx.x) * 8;
    *(v8bf*)(wdst + (size_t)which * HD * ND + off) = cvt8(W + off);
}

// ---------------------------------------------------------------------------
// Kernel 1: fused projections. grid = (ROWS/128, 3), block = 256 (8 waves).
//  y==0: qdst = x @ Wq^T + bq ; y==1: kdst = buffer @ Wk^T + bk ;
//  y==2: vdst = buffer @ Wv^T + bv.  Output bf16 [16384, 64].
// B-fragments come straight from the pre-converted bf16 weights.
// ---------------------------------------------------------------------------
__global__ __launch_bounds__(256)
void proj_kernel(const float* __restrict__ x, const float* __restrict__ buf,
                 const __bf16* __restrict__ wbf,
                 const float* __restrict__ bq, const float* __restrict__ bk,
                 const float* __restrict__ bv,
                 __bf16* __restrict__ qdst, __bf16* __restrict__ kdst,
                 __bf16* __restrict__ vdst) {
    const int which = blockIdx.y;
    const float*  src  = (which == 0) ? x  : buf;
    const __bf16* W    = wbf + (size_t)which * HD * ND;
    const float*  bias = (which == 0) ? bq : (which == 1) ? bk : bv;
    __bf16*       dst  = (which == 0) ? qdst : (which == 1) ? kdst : vdst;

    const int wave = threadIdx.x >> 5;
    const int lane = threadIdx.x & 31;
    const int m    = lane & 15;          // row (A) / col (B,C) index for this lane
    const int half = lane >> 4;
    const int koff1 = half ? 8 : 0;      // element 0..7 K-offset (16-bit A/B layout)
    const int koff2 = koff1 + 16;        // element 8..15 K-offset

    const size_t row = (size_t)blockIdx.x * 128 + wave * 16 + m;

    v8f acc[4];
    #pragma unroll
    for (int nt = 0; nt < 4; ++nt) acc[nt] = (v8f){0,0,0,0,0,0,0,0};

    #pragma unroll 4
    for (int d0 = 0; d0 < ND; d0 += 32) {
        const float* arow = src + row * ND + d0;
        v16bf afrag = cat16(cvt8(arow + koff1), cvt8(arow + koff2));
        #pragma unroll
        for (int nt = 0; nt < 4; ++nt) {
            const __bf16* wrow = W + (size_t)(nt * 16 + m) * ND + d0;
            v16bf bfrag = frag_row(wrow, koff1, koff2);
            acc[nt] = wmma_bf16(afrag, bfrag, acc[nt]);
        }
    }

    // epilogue: + bias, cvt bf16, store. C layout: elem j -> row j+8*half, col nt*16+m
    const size_t rbase = (size_t)blockIdx.x * 128 + wave * 16;
    #pragma unroll
    for (int nt = 0; nt < 4; ++nt) {
        const int col = nt * 16 + m;
        const float bv_ = bias[col];
        #pragma unroll
        for (int j = 0; j < 8; ++j) {
            size_t r = rbase + j + 8 * half;
            dst[r * HD + col] = cvt1(acc[nt][j] + bv_);
        }
    }
}

// ---------------------------------------------------------------------------
// Kernel 2: flash attention. grid = B*(Q/64) = 256, block = 128 (4 waves).
// Each wave: 16 Q-rows. Per iteration: 64-key tile + 64x64 mask tile in LDS.
// Softmax runs in the exp2 domain (logits pre-scaled by log2 e) so every
// exponential is a bare v_exp_f32.
// ---------------------------------------------------------------------------
__global__ __launch_bounds__(128)
void attn_kernel(const __bf16* __restrict__ qbuf, const __bf16* __restrict__ kbuf,
                 const __bf16* __restrict__ vbuf, const int* __restrict__ mask,
                 float* __restrict__ out) {
    __shared__ __align__(16) __bf16 Ktile[64 * HD];       // [key][ch]
    __shared__ __align__(16) __bf16 Vtile[HD * 64];       // [ch][key] (transposed)
    __shared__ __align__(16) __bf16 Ptile[4][16 * 64];    // per-wave [row][key]
    __shared__ __align__(16) int    Mtile[64 * 64];       // [qrow][key] mask ints

    const int b  = blockIdx.x >> 6;
    const int q0 = (blockIdx.x & 63) * 64;

    const int wave = threadIdx.x >> 5;
    const int lane = threadIdx.x & 31;
    const int m    = lane & 15;
    const int half = lane >> 4;
    const int koff1 = half ? 8 : 0;
    const int koff2 = koff1 + 16;

    // Q fragments: rows q0 + wave*16 + m, held in registers for whole K loop
    const __bf16* qrow = qbuf + ((size_t)b * NQ + q0 + wave * 16 + m) * HD;
    v16bf qf[2];
    #pragma unroll
    for (int s = 0; s < 2; ++s)
        qf[s] = frag_row(qrow, s * 32 + koff1, s * 32 + koff2);

    v8f o[4];
    #pragma unroll
    for (int nt = 0; nt < 4; ++nt) o[nt] = (v8f){0,0,0,0,0,0,0,0};
    float mrun[8], lrun[8];            // running max / sum, exp2 domain
    #pragma unroll
    for (int j = 0; j < 8; ++j) { mrun[j] = -3.0e38f; lrun[j] = 0.0f; }

    const size_t kvbase = (size_t)b * NK;
    const size_t mbase  = ((size_t)b * NQ + q0) * (size_t)NK;   // mask tile origin

    #pragma unroll 1
    for (int k0 = 0; k0 < NK; k0 += 64) {
        // ---- stage K tile [64 keys][64 ch] and V tile transposed [ch][key] ----
        #pragma unroll
        for (int c = 0; c < 4; ++c) {
            int chunk = c * 128 + threadIdx.x;       // 512 chunks of 8 bf16
            int key = chunk >> 3;
            int cho = (chunk & 7) * 8;
            const __bf16* ks = kbuf + (kvbase + k0 + key) * HD + cho;
            *(v8bf*)(Ktile + key * HD + cho) = *(const v8bf*)ks;
            v8bf vv = *(const v8bf*)(vbuf + (kvbase + k0 + key) * HD + cho);
            #pragma unroll
            for (int i = 0; i < 8; ++i)
                Vtile[(cho + i) * 64 + key] = vv[i];
        }
        // ---- stage mask tile [64 qrows][64 keys], fully coalesced b128 ----
        #pragma unroll
        for (int c = 0; c < 8; ++c) {
            int chunk = c * 128 + threadIdx.x;       // 1024 chunks of 4 ints
            int qr = chunk >> 4;
            int kc = (chunk & 15) * 4;
            *(int4*)(Mtile + qr * 64 + kc) =
                *(const int4*)(mask + mbase + (size_t)qr * NK + k0 + kc);
        }
        if (k0 + 64 < NK)
            __builtin_prefetch(kbuf + (kvbase + k0 + 64) * HD, 0, 1);
        __syncthreads();

        // ---- S = Q * K^T (accumulate f32) ----
        v8f s[4];
        #pragma unroll
        for (int nt = 0; nt < 4; ++nt) s[nt] = (v8f){0,0,0,0,0,0,0,0};
        #pragma unroll
        for (int ks = 0; ks < 2; ++ks) {
            #pragma unroll
            for (int nt = 0; nt < 4; ++nt) {
                const __bf16* kp = Ktile + (nt * 16 + m) * HD + ks * 32;
                v16bf bf = frag_row(kp, koff1, koff2);
                s[nt] = wmma_bf16(qf[ks], bf, s[nt]);
            }
        }

        // ---- mask + scale + online softmax (exp2 domain); P (bf16) -> LDS ----
        #pragma unroll
        for (int j = 0; j < 8; ++j) {
            const int tr = wave * 16 + j + 8 * half;   // q-row within block tile
            const int* mrow = Mtile + tr * 64;
            float sv[4];
            float tmax = -3.0e38f;
            #pragma unroll
            for (int t = 0; t < 4; ++t) {
                float v = s[t][j] * SCALE2;            // logit * log2(e)
                v = mrow[t * 16 + m] ? v : FILL2;
                sv[t] = v;
                tmax = fmaxf(tmax, v);
            }
            // row-wide max: xor masks 1,2,4,8 stay within the 16-lane row group
            #pragma unroll
            for (int d = 1; d <= 8; d <<= 1)
                tmax = fmaxf(tmax, __shfl_xor(tmax, d, 32));
            float mnew  = fmaxf(mrun[j], tmax);
            float alpha = exp2f(mrun[j] - mnew);       // bare v_exp_f32
            mrun[j] = mnew;
            float ps = 0.0f;
            #pragma unroll
            for (int t = 0; t < 4; ++t) {
                float p = exp2f(sv[t] - mnew);         // bare v_exp_f32
                Ptile[wave][(j + 8 * half) * 64 + t * 16 + m] = cvt1(p);
                ps += p;
            }
            #pragma unroll
            for (int d = 1; d <= 8; d <<= 1)
                ps += __shfl_xor(ps, d, 32);
            lrun[j] = lrun[j] * alpha + ps;
            #pragma unroll
            for (int nt = 0; nt < 4; ++nt) o[nt][j] *= alpha;
        }

        // ---- O += P * V (P A-frags from LDS; V B-frags from transposed tile) ----
        #pragma unroll
        for (int ks = 0; ks < 2; ++ks) {
            const __bf16* pp = Ptile[wave] + m * 64 + ks * 32;
            v16bf pf = frag_row(pp, koff1, koff2);
            #pragma unroll
            for (int nt = 0; nt < 4; ++nt) {
                const __bf16* vp = Vtile + (nt * 16 + m) * 64 + ks * 32;
                v16bf vf = frag_row(vp, koff1, koff2);
                o[nt] = wmma_bf16(pf, vf, o[nt]);
            }
        }
        __syncthreads();   // K/V/M tiles reused next iteration
    }

    // ---- epilogue: normalize by l, store f32 output [B, Q, 64] ----
    #pragma unroll
    for (int nt = 0; nt < 4; ++nt) {
        const int col = nt * 16 + m;
        #pragma unroll
        for (int j = 0; j < 8; ++j) {
            const int qidx = q0 + wave * 16 + j + 8 * half;
            out[((size_t)b * NQ + qidx) * HD + col] = o[nt][j] / lrun[j];
        }
    }
}

// ---------------------------------------------------------------------------
// Host entry
// ---------------------------------------------------------------------------
extern "C" void kernel_launch(void* const* d_in, const int* in_sizes, int n_in,
                              void* d_out, int out_size, void* d_ws, size_t ws_size,
                              hipStream_t stream) {
    const float* x   = (const float*)d_in[0];   // [4,4096,512]
    const float* buf = (const float*)d_in[1];   // [4,4096,512]
    const int*   msk = (const int*)  d_in[2];   // [4,4096,4096]
    const float* Wq  = (const float*)d_in[3];   // [64,512]
    const float* bq  = (const float*)d_in[4];
    const float* Wk  = (const float*)d_in[5];
    const float* bk  = (const float*)d_in[6];
    const float* Wv  = (const float*)d_in[7];
    const float* bv  = (const float*)d_in[8];
    float* out = (float*)d_out;                 // [4,4096,64]

    const size_t rows = (size_t)NB * NQ;        // 16384
    __bf16* qb = (__bf16*)d_ws;                 // [16384,64] bf16
    __bf16* kb = qb + rows * HD;
    __bf16* vb = kb + rows * HD;
    __bf16* wb = vb + rows * HD;                // 3 x [64,512] bf16 weights
    (void)in_sizes; (void)n_in; (void)out_size; (void)ws_size;

    wcvt_kernel<<<dim3(3 * (HD * ND) / (256 * 8)), 256, 0, stream>>>(Wq, Wk, Wv, wb);

    dim3 pgrid(rows / 128, 3);
    proj_kernel<<<pgrid, 256, 0, stream>>>(x, buf, wb, bq, bk, bv, qb, kb, vb);

    dim3 agrid(NB * (NQ / 64));
    attn_kernel<<<agrid, 128, 0, stream>>>(qb, kb, vb, msk, out);
}
